// AetherAttention_37718402793750
// MI455X (gfx1250) — matrix operand order repaired and checked by reading, not compiled, not run on hardware
//
#include <hip/hip_runtime.h>
#include <hip/hip_bf16.h>
#include <math.h>

typedef __attribute__((ext_vector_type(16))) __bf16 v16bf;
typedef __attribute__((ext_vector_type(8)))  float  v8f;

#define B_    2
#define H_    8
#define MSEQ  2048
#define NSEQ  2048
#define DH    64
#define BLK   64
#define NQB   32
#define NKB   32
#define SCALE 0.125f
#define LOGTHR (-1.8971199848858813f)
#define LDSS  72   // LDS row stride in bf16 elements

// ---------- helpers ----------------------------------------------------------

// float -> bf16 via native conversion (gfx1250 v_cvt_*bf16_f32, RNE)
__device__ __forceinline__ __bf16 f2bf(float f) { return (__bf16)f; }

// order-preserving float<->uint encode (atomicMax on floats incl. negatives)
__device__ __forceinline__ unsigned fenc(float f) {
  unsigned u = __float_as_uint(f);
  return (u & 0x80000000u) ? ~u : (u | 0x80000000u);
}
__device__ __forceinline__ float fdec(unsigned k) {
  unsigned u = (k & 0x80000000u) ? (k & 0x7FFFFFFFu) : ~k;
  return __uint_as_float(u);
}

// lane-xor data exchange: DPP8 for xor within 8-lane groups, ds_swizzle for xor8
__device__ __forceinline__ constexpr int dpp8_xor_sel(int m) {
  int s = 0;
  for (int i = 0; i < 8; ++i) s |= ((i ^ m) & 7) << (3 * i);
  return s;
}
template <int M>
__device__ __forceinline__ float lane_xor_f(float x) {
  if constexpr (M < 8) {
    return __int_as_float(__builtin_amdgcn_mov_dpp8(__float_as_int(x), dpp8_xor_sel(M)));
  } else {
    // group-of-32 swizzle: and=0x1F, or=0, xor=M  -> src_lane = lane ^ M
    return __int_as_float(__builtin_amdgcn_ds_swizzle(__float_as_int(x), (M << 10) | 0x1F));
  }
}
__device__ __forceinline__ float red16_max(float x) {
  x = fmaxf(x, lane_xor_f<1>(x));
  x = fmaxf(x, lane_xor_f<2>(x));
  x = fmaxf(x, lane_xor_f<4>(x));
  x = fmaxf(x, lane_xor_f<8>(x));
  return x;
}
__device__ __forceinline__ float red16_sum(float x) {
  x += lane_xor_f<1>(x);
  x += lane_xor_f<2>(x);
  x += lane_xor_f<4>(x);
  x += lane_xor_f<8>(x);
  return x;
}

// A fragment, 16x32 bf16 (ISA 7.12.2): row m = lane%16;
// elems 0..7  -> K = kk + 8*half + e ; elems 8..15 -> K = kk + 16 + 8*half + (e-8)
__device__ __forceinline__ v16bf load_a_frag(const __bf16* row, int half, int kk) {
  v16bf a;
  const __bf16* p0 = row + kk + 8 * half;
  const __bf16* p1 = row + kk + 16 + 8 * half;
#pragma unroll
  for (int e = 0; e < 8; ++e) { a[e] = p0[e]; a[8 + e] = p1[e]; }
  return a;
}
// B fragment, 32x16 bf16: col n = lane%16; elems 0..7 -> K = kk + 8*half + e ; elems 8..15 -> K = kk + 16 + 8*half + (e-8)
// caller passes row = &LDS[n][0] where LDS[n][k] == B[k][n]
__device__ __forceinline__ v16bf load_b_frag(const __bf16* row, int half, int kk) {
  v16bf b;
  const __bf16* p0 = row + kk + 8 * half;
  const __bf16* p1 = row + kk + 16 + 8 * half;
#pragma unroll
  for (int e = 0; e < 8; ++e) { b[e] = p0[e]; b[8 + e] = p1[e]; }
  return b;
}

#define WMMA_BF16(A, Bv, C) \
  __builtin_amdgcn_wmma_f32_16x16x32_bf16(false, (A), false, (Bv), (short)0, (C), false, false)

// ---------- phase 1: key-block centroids + radii (f32, matches reference) ----

__global__ __launch_bounds__(64) void aether_geom(const float* __restrict__ kin,
                                                  float* __restrict__ cen,
                                                  float* __restrict__ rad) {
  int jb = blockIdx.x & (NKB - 1);
  int h  = (blockIdx.x >> 5) & (H_ - 1);
  int b  = blockIdx.x >> 8;
  int t  = threadIdx.x;
  __shared__ float cS[DH];
  __shared__ float red[BLK];
  const float* kb = kin + ((size_t)(b * NSEQ + jb * BLK) * H_ + h) * DH;

  float s = 0.f;
#pragma unroll 8
  for (int n = 0; n < BLK; ++n) s += kb[(size_t)n * (H_ * DH) + t];
  float c = s * (1.0f / (float)BLK);
  cS[t] = c;
  cen[((size_t)(b * H_ + h) * NKB + jb) * DH + t] = c;
  __syncthreads();

  float d2 = 0.f;
#pragma unroll 8
  for (int d = 0; d < DH; ++d) {
    float diff = kb[(size_t)t * (H_ * DH) + d] - cS[d];
    d2 += diff * diff;
  }
  red[t] = d2;
  __syncthreads();
  for (int w = 32; w > 0; w >>= 1) {
    if (t < w) red[t] = fmaxf(red[t], red[t + w]);
    __syncthreads();
  }
  if (t == 0) rad[(size_t)(b * H_ + h) * NKB + jb] = sqrtf(red[0]);
}

// ---------- phase 2: per-query-block keep bitmask (f32) ----------------------

__global__ __launch_bounds__(64) void aether_mask(const float* __restrict__ q,
                                                  const float* __restrict__ cen,
                                                  const float* __restrict__ rad,
                                                  unsigned* __restrict__ mask) {
  int ib = blockIdx.x & (NQB - 1);
  int h  = (blockIdx.x >> 5) & (H_ - 1);
  int b  = blockIdx.x >> 8;
  int t  = threadIdx.x;
  int bh = b * H_ + h;

  __shared__ float    cS[NKB * DH];
  __shared__ unsigned qcb[NKB];
  __shared__ unsigned qnb;
  __shared__ unsigned mk;

  for (int i = t; i < NKB * DH; i += 64) cS[i] = cen[(size_t)bh * NKB * DH + i];
  if (t < NKB) qcb[t] = 0u;          // encoded -inf
  if (t == 0) { qnb = 0u; mk = 0u; }
  __syncthreads();

  const float* qr = q + ((size_t)(b * MSEQ + ib * BLK + t) * H_ + h) * DH;
  float qv[DH];
  float n2 = 0.f;
#pragma unroll
  for (int d = 0; d < DH; ++d) { qv[d] = qr[d]; n2 += qv[d] * qv[d]; }
  atomicMax(&qnb, fenc(sqrtf(n2)));
  for (int j = 0; j < NKB; ++j) {
    float dot = 0.f;
#pragma unroll
    for (int d = 0; d < DH; ++d) dot += qv[d] * cS[j * DH + d];
    atomicMax(&qcb[j], fenc(dot));
  }
  __syncthreads();

  if (t < NKB) {  // lanes 0..31 = first wave32
    float ub = SCALE * (fdec(qcb[t]) + fdec(qnb) * rad[(size_t)bh * NKB + t]);
    float mx = ub;
#pragma unroll
    for (int off = 1; off <= 16; off <<= 1) mx = fmaxf(mx, __shfl_xor(mx, off, 32));
    if (ub >= mx + LOGTHR) atomicOr(&mk, 1u << t);
  }
  __syncthreads();
  if (t == 0) mask[(size_t)bh * NQB + ib] = mk;
}

// ---------- phase 3: block-sparse flash attention, bf16 WMMA -----------------
// 4 waves/WG, one WG per (b,h,query-block). LDS double-buffered K/V with
// register staging; Q A-fragments hoisted to registers; P strip reuses Q LDS.

__global__ __launch_bounds__(128) void aether_attn(const float* __restrict__ q,
                                                   const float* __restrict__ kg,
                                                   const float* __restrict__ vg,
                                                   const unsigned* __restrict__ maskbuf,
                                                   float* __restrict__ out) {
  int ib = blockIdx.x & (NQB - 1);
  int h  = (blockIdx.x >> 5) & (H_ - 1);
  int b  = blockIdx.x >> 8;
  int tid  = threadIdx.x;
  int wave = tid >> 5;
  int lane = tid & 31;
  int half = lane >> 4;
  int l16  = lane & 15;

  __shared__ __bf16 QPs[BLK][LDSS];      // Q tile; later reused as P strips
  __shared__ __bf16 KsB[2][BLK][LDSS];   // Ks[n][k] = K row n
  __shared__ __bf16 VTB[2][DH][LDSS];    // VT[d][t] = V[t][d]

  const size_t rowstep = (size_t)(H_ * DH / 4);  // float4 stride between tokens

  // ---- Q tile -> LDS (pre-scaled by 2^-3, exact in bf16) ----
  {
    const float4* qp = (const float4*)(q + ((size_t)(b * MSEQ + ib * BLK) * H_ + h) * DH);
#pragma unroll
    for (int r = 0; r < 8; ++r) {
      int i = tid + 128 * r, m = i >> 4, c4 = i & 15;
      float4 f = qp[(size_t)m * rowstep + c4];
      QPs[m][c4 * 4 + 0] = f2bf(f.x * SCALE);
      QPs[m][c4 * 4 + 1] = f2bf(f.y * SCALE);
      QPs[m][c4 * 4 + 2] = f2bf(f.z * SCALE);
      QPs[m][c4 * 4 + 3] = f2bf(f.w * SCALE);
    }
  }

  unsigned kmask = maskbuf[(size_t)(b * H_ + h) * NQB + ib];  // uniform scalar
  unsigned mrem  = kmask;                                     // always != 0

  // ---- stage first kept block into regs, then LDS buffer 0 ----
  float4 kf[8], vf[8];
  {
    int j0 = __builtin_ctz(mrem); mrem &= mrem - 1;
    const float4* kp = (const float4*)(kg + ((size_t)(b * NSEQ + j0 * BLK) * H_ + h) * DH);
    const float4* vp = (const float4*)(vg + ((size_t)(b * NSEQ + j0 * BLK) * H_ + h) * DH);
#pragma unroll
    for (int r = 0; r < 8; ++r) {
      int i = tid + 128 * r;
      kf[r] = kp[(size_t)(i >> 4) * rowstep + (i & 15)];
      vf[r] = vp[(size_t)(i >> 4) * rowstep + (i & 15)];
    }
#pragma unroll
    for (int r = 0; r < 8; ++r) {
      int i = tid + 128 * r, t = i >> 4, c4 = i & 15;
      KsB[0][t][c4 * 4 + 0] = f2bf(kf[r].x);
      KsB[0][t][c4 * 4 + 1] = f2bf(kf[r].y);
      KsB[0][t][c4 * 4 + 2] = f2bf(kf[r].z);
      KsB[0][t][c4 * 4 + 3] = f2bf(kf[r].w);
      VTB[0][c4 * 4 + 0][t] = f2bf(vf[r].x);
      VTB[0][c4 * 4 + 1][t] = f2bf(vf[r].y);
      VTB[0][c4 * 4 + 2][t] = f2bf(vf[r].z);
      VTB[0][c4 * 4 + 3][t] = f2bf(vf[r].w);
    }
  }
  __syncthreads();

  // ---- hoist loop-invariant Q A-fragments into registers ----
  const __bf16* qrow = &QPs[wave * 16 + l16][0];
  v16bf qa0 = load_a_frag(qrow, half, 0);
  v16bf qa1 = load_a_frag(qrow, half, 32);
  const __bf16* prow = qrow;  // P strip overwrites this wave's own Q rows

  v8f acc[4] = {};
  float rowmax[8], rowsum[8];
#pragma unroll
  for (int g = 0; g < 8; ++g) { rowmax[g] = -__builtin_inff(); rowsum[g] = 0.f; }

  int cur = 0;
  for (;;) {
    bool more = (mrem != 0);  // uniform
    if (more) {               // issue next block's global loads before compute
      int jn = __builtin_ctz(mrem); mrem &= mrem - 1;
      const float4* kp = (const float4*)(kg + ((size_t)(b * NSEQ + jn * BLK) * H_ + h) * DH);
      const float4* vp = (const float4*)(vg + ((size_t)(b * NSEQ + jn * BLK) * H_ + h) * DH);
#pragma unroll
      for (int r = 0; r < 8; ++r) {
        int i = tid + 128 * r;
        kf[r] = kp[(size_t)(i >> 4) * rowstep + (i & 15)];
        vf[r] = vp[(size_t)(i >> 4) * rowstep + (i & 15)];
      }
    }

    const __bf16(*ksb)[LDSS] = KsB[cur];
    const __bf16(*vtb)[LDSS] = VTB[cur];

    // ---- S strip = Qstrip * K^T (pre-scaled): 4 tiles x 2 WMMAs ----
    float S[4][8];
#pragma unroll
    for (int nt = 0; nt < 4; ++nt) {
      v8f c = {};
      v16bf b0 = load_b_frag(&ksb[nt * 16 + l16][0], half, 0);
      c = WMMA_BF16(qa0, b0, c);
      v16bf b1 = load_b_frag(&ksb[nt * 16 + l16][0], half, 32);
      c = WMMA_BF16(qa1, b1, c);
#pragma unroll
      for (int g = 0; g < 8; ++g) S[nt][g] = c[g];
    }

    // ---- online softmax (row g + 8*half, spread over 16 lanes of the half) --
#pragma unroll
    for (int g = 0; g < 8; ++g) {
      float mx = red16_max(fmaxf(fmaxf(S[0][g], S[1][g]), fmaxf(S[2][g], S[3][g])));
      float nm    = fmaxf(rowmax[g], mx);
      float alpha = __expf(rowmax[g] - nm);
      rowmax[g]   = nm;
      float s = 0.f;
#pragma unroll
      for (int nt = 0; nt < 4; ++nt) {
        float p = __expf(S[nt][g] - nm);
        S[nt][g] = p;
        s += p;
      }
      rowsum[g] = rowsum[g] * alpha + red16_sum(s);
#pragma unroll
      for (int nt = 0; nt < 4; ++nt) acc[nt][g] *= alpha;
    }

    // ---- C-layout -> A-layout via this wave's own LDS strip (DS in-order) ---
#pragma unroll
    for (int nt = 0; nt < 4; ++nt)
#pragma unroll
      for (int g = 0; g < 8; ++g)
        QPs[wave * 16 + g + 8 * half][nt * 16 + l16] = f2bf(S[nt][g]);

    v16bf pa0 = load_a_frag(prow, half, 0);
    v16bf pa1 = load_a_frag(prow, half, 32);

    // ---- O += P * V: 4 tiles x 2 WMMAs ----
#pragma unroll
    for (int dt = 0; dt < 4; ++dt) {
      v16bf b0 = load_b_frag(&vtb[dt * 16 + l16][0], half, 0);
      acc[dt] = WMMA_BF16(pa0, b0, acc[dt]);
      v16bf b1 = load_b_frag(&vtb[dt * 16 + l16][0], half, 32);
      acc[dt] = WMMA_BF16(pa1, b1, acc[dt]);
    }

    if (!more) break;
    __syncthreads();                 // all waves done reading buffer cur^1
    cur ^= 1;
#pragma unroll
    for (int r = 0; r < 8; ++r) {    // drain staged regs into the other buffer
      int i = tid + 128 * r, t = i >> 4, c4 = i & 15;
      KsB[cur][t][c4 * 4 + 0] = f2bf(kf[r].x);
      KsB[cur][t][c4 * 4 + 1] = f2bf(kf[r].y);
      KsB[cur][t][c4 * 4 + 2] = f2bf(kf[r].z);
      KsB[cur][t][c4 * 4 + 3] = f2bf(kf[r].w);
      VTB[cur][c4 * 4 + 0][t] = f2bf(vf[r].x);
      VTB[cur][c4 * 4 + 1][t] = f2bf(vf[r].y);
      VTB[cur][c4 * 4 + 2][t] = f2bf(vf[r].z);
      VTB[cur][c4 * 4 + 3][t] = f2bf(vf[r].w);
    }
    __syncthreads();                 // make new buffer visible
  }

  // ---- epilogue: normalize and store ----
#pragma unroll
  for (int g = 0; g < 8; ++g) {
    float inv = 1.0f / rowsum[g];
    int m = ib * BLK + wave * 16 + g + 8 * half;
    float* op = out + ((size_t)(b * MSEQ + m) * H_ + h) * DH;
#pragma unroll
    for (int dt = 0; dt < 4; ++dt) op[dt * 16 + l16] = acc[dt][g] * inv;
  }
}

// ---------- launch -----------------------------------------------------------

extern "C" void kernel_launch(void* const* d_in, const int* in_sizes, int n_in,
                              void* d_out, int out_size, void* d_ws, size_t ws_size,
                              hipStream_t stream) {
  const float* q = (const float*)d_in[0];
  const float* k = (const float*)d_in[1];
  const float* v = (const float*)d_in[2];
  float* out = (float*)d_out;

  // workspace: centroids [B*H*NKB*DH] f32 | radii [B*H*NKB] f32 | masks [B*H*NQB] u32
  float*    cen  = (float*)d_ws;
  float*    rad  = cen + (size_t)B_ * H_ * NKB * DH;
  unsigned* mask = (unsigned*)(rad + (size_t)B_ * H_ * NKB);

  aether_geom<<<dim3(B_ * H_ * NKB), 64, 0, stream>>>(k, cen, rad);
  aether_mask<<<dim3(B_ * H_ * NQB), 64, 0, stream>>>(q, cen, rad, mask);
  aether_attn<<<dim3(B_ * H_ * NQB), 128, 0, stream>>>(q, k, v, mask, out);
}
